// CoTracker2_5334349382125
// MI455X (gfx1250) — matrix operand order repaired and checked by reading, not compiled, not run on hardware
//
#include <hip/hip_runtime.h>
#include <hip/hip_bf16.h>

typedef __bf16 bf16;
typedef __attribute__((ext_vector_type(16))) __bf16 v16bf;
typedef __attribute__((ext_vector_type(8)))  __bf16 v8bf;
typedef __attribute__((ext_vector_type(8)))  float  v8f;
typedef __attribute__((ext_vector_type(2)))  float  v2f;

#define S_    16
#define N_    512
#define C_    64
#define MTOK  8192      // N_*S_ token rows, row = n*16 + s
#define XK    224       // 196 padded to 7*32 for K of bf16 WMMA
#define QLD   192
#define FLD   80        // flow head: 66 padded to 5*16

// ---------------- pyramid ----------------
__global__ void k_avgpool(const float* __restrict__ in, float* __restrict__ out,
                          int h, int w) {
  int idx = blockIdx.x * blockDim.x + threadIdx.x;
  int total = S_ * C_ * h * w;
  if (idx >= total) return;
  int x = idx % w, y = (idx / w) % h, sc = idx / (w * h);
  int W2 = w * 2;
  const float* p = in + ((size_t)sc * (h * 2) + y * 2) * W2 + x * 2;
  out[idx] = 0.25f * (p[0] + p[1] + p[W2] + p[W2 + 1]);
}

// ---------------- weight f32 -> bf16, TRANSPOSED (dout x dinP), zero padded ------
// WT[c][r] = W[r][c]; contiguous K per output column -> b128 loads in the GEMM
__global__ void k_padw(const float* __restrict__ src, bf16* __restrict__ dst,
                       int din, int dout, int dinP, int doutP) {
  int idx = blockIdx.x * blockDim.x + threadIdx.x;
  if (idx >= dinP * doutP) return;
  int r = idx % dinP, c = idx / dinP;   // dst row-major (doutP x dinP)
  float v = (r < din && c < dout) ? src[r * dout + c] : 0.0f;
  dst[idx] = (bf16)v;
}

__global__ void k_f2bf(const float* __restrict__ src, bf16* __restrict__ dst, int n) {
  int i = blockIdx.x * blockDim.x + threadIdx.x;
  if (i < n) dst[i] = (bf16)src[i];
}

// ---------------- direct correlation sampling ----------------
// one wave32 per (s,n): dot(track_feat, fmap pixel) at 25 bilinear taps x 4 levels
__global__ void k_corr(const float* __restrict__ coords, const float* __restrict__ tfeat,
                       const float* __restrict__ p0, const float* __restrict__ p1,
                       const float* __restrict__ p2, const float* __restrict__ p3,
                       bf16* __restrict__ xbf) {
  int wid  = (blockIdx.x * blockDim.x + threadIdx.x) >> 5;
  int lane = threadIdx.x & 31;
  if (wid >= S_ * N_) return;
  int s = wid >> 9, n = wid & 511;
  float t0 = tfeat[((size_t)s * N_ + n) * C_ + lane];
  float t1 = tfeat[((size_t)s * N_ + n) * C_ + lane + 32];
  float px = coords[(s * N_ + n) * 2 + 0];
  float py = coords[(s * N_ + n) * 2 + 1];
  const float* pyr[4] = {p0, p1, p2, p3};
  bf16* dst = xbf + (size_t)(n * S_ + s) * XK + 96;
  for (int lvl = 0; lvl < 4; ++lvl) {
    int hw = 64 >> lvl;
    const float* fm = pyr[lvl] + (size_t)s * C_ * hw * hw;
    const float* f0 = fm + (size_t)lane * hw * hw;
    const float* f1 = fm + (size_t)(lane + 32) * hw * hw;
    float inv = 1.0f / (float)(1 << lvl);
    float cx = px * inv, cy = py * inv;
    __builtin_prefetch(f0 + (int)cy * hw, 0, 1);   // global_prefetch_b8
    __builtin_prefetch(f1 + (int)cy * hw, 0, 1);
    for (int k = 0; k < 25; ++k) {
      float sx = cx + (float)(k / 5 - 2);          // offx = d[i] (meshgrid 'ij')
      float sy = cy + (float)(k % 5 - 2);
      sx = fminf(fmaxf(sx, 0.f), (float)(hw - 1));
      sy = fminf(fmaxf(sy, 0.f), (float)(hw - 1));
      float fx0 = floorf(sx), fy0 = floorf(sy);
      int x0 = (int)fx0, y0 = (int)fy0;
      int x1 = min(x0 + 1, hw - 1), y1 = min(y0 + 1, hw - 1);
      float wx = sx - fx0, wy = sy - fy0;
      float b0 = f0[y0 * hw + x0] * (1 - wy) * (1 - wx) + f0[y0 * hw + x1] * (1 - wy) * wx
               + f0[y1 * hw + x0] * wy * (1 - wx)       + f0[y1 * hw + x1] * wy * wx;
      float b1 = f1[y0 * hw + x0] * (1 - wy) * (1 - wx) + f1[y0 * hw + x1] * (1 - wy) * wx
               + f1[y1 * hw + x0] * wy * (1 - wx)       + f1[y1 * hw + x1] * wy * wx;
      float acc = t0 * b0 + t1 * b1;
      for (int d = 16; d >= 1; d >>= 1) acc += __shfl_xor(acc, d, 32);
      if (lane == 0) dst[lvl * 25 + k] = (bf16)(acc * 0.125f); // 1/sqrt(64)
    }
  }
}

// ---------------- build x = [flow_emb(32) | tf(64) | fcorr(100) | pad] ----------------
__global__ void k_buildx(const float* __restrict__ coords, const float* __restrict__ tfeat,
                         bf16* __restrict__ xbf) {
  int row = blockIdx.x * blockDim.x + threadIdx.x;  // n*16+s
  if (row >= MTOK) return;
  int s = row & 15, n = row >> 4;
  bf16* dst = xbf + (size_t)row * XK;
  float fx = coords[(s * N_ + n) * 2 + 0] - coords[n * 2 + 0];
  float fy = coords[(s * N_ + n) * 2 + 1] - coords[n * 2 + 1];
  for (int j = 0; j < 8; ++j) {
    float ax = fx * (125.0f * j);
    dst[2 * j]     = (bf16)sinf(ax);
    dst[2 * j + 1] = (bf16)cosf(ax);
    float ay = fy * (125.0f * j);
    dst[16 + 2 * j]     = (bf16)sinf(ay);
    dst[16 + 2 * j + 1] = (bf16)cosf(ay);
  }
  const float* tf = tfeat + ((size_t)s * N_ + n) * C_;
  for (int c = 0; c < 64; ++c) dst[32 + c] = (bf16)tf[c];
  for (int c = 196; c < XK; ++c) dst[c] = (bf16)0.0f;
}

// ---------------- bf16 WMMA GEMM with fused bias/act/residual ----------------
// one wave per 16x16 output tile; v_wmma_f32_16x16x32_bf16 over K.
// A row-major (M x lda), WT transposed (doutP x K). All operand fetches are b128.
__global__ void k_gemm(const bf16* __restrict__ A, int lda,
                       const bf16* __restrict__ WT, int ldw,   // ldw = doutP (out pitch)
                       const float* __restrict__ bias, int dout,
                       const float* __restrict__ res,
                       float* __restrict__ outf, bf16* __restrict__ outbf,
                       int M, int K, int act, int tfmode) {
  int wid = (blockIdx.x * blockDim.x + threadIdx.x) >> 5;
  int lane = threadIdx.x & 31;
  int half = lane >> 4, lm = lane & 15;
  int tilesN = ldw >> 4;
  int tm = wid / tilesN, tn = wid % tilesN;
  if (tm * 16 >= M) return;
  int m0 = tm * 16, n0 = tn * 16;
  v8f acc = {0.f, 0.f, 0.f, 0.f, 0.f, 0.f, 0.f, 0.f};
  const bf16* arow = A  + (size_t)(m0 + lm) * lda + half * 8;   // 16B aligned
  const bf16* brow = WT + (size_t)(n0 + lm) * K   + half * 8;   // 16B aligned
  for (int k0 = 0; k0 < K; k0 += 32) {
    // A 16x32 (ISA §7.12.2): lane needs K = [k0+8h .. +7] and [k0+16+8h .. +7]
    v8bf alo = *(const v8bf*)(arow + k0);
    v8bf ahi = *(const v8bf*)(arow + k0 + 16);
    v16bf a = __builtin_shufflevector(alo, ahi, 0, 1, 2, 3, 4, 5, 6, 7,
                                                8, 9, 10, 11, 12, 13, 14, 15);
    // B 32x16: lane holds column n0+lm, K = [k0+8h .. +7] and [k0+16+8h .. +7]
    v8bf blo = *(const v8bf*)(brow + k0);
    v8bf bhi = *(const v8bf*)(brow + k0 + 16);
    v16bf b = __builtin_shufflevector(blo, bhi, 0, 1, 2, 3, 4, 5, 6, 7,
                                                8, 9, 10, 11, 12, 13, 14, 15);
    acc = __builtin_amdgcn_wmma_f32_16x16x32_bf16(false, a, false, b, (short)0, acc,
                                                  false, false);
  }
  int n = n0 + lm;
  float bv = (n < dout) ? bias[n] : 0.0f;
#pragma unroll
  for (int v = 0; v < 8; ++v) {
    int m = m0 + v + half * 8;
    float val = acc[v] + bv;
    if (act == 1) {           // gelu tanh approx
      float x3 = val * val * val;
      val = 0.5f * val * (1.0f + tanhf(0.7978845608f * (val + 0.044715f * x3)));
    } else if (act == 2) {    // gelu exact
      val = 0.5f * val * (1.0f + erff(val * 0.70710678f));
    }
    size_t oi;
    if (tfmode) { int ss = m & 15, nn = m >> 4; oi = ((size_t)ss * N_ + nn) * C_ + n; }
    else        { oi = (size_t)m * ldw + n; }
    if (res) val += res[oi];
    outf[oi] = val;
    if (outbf) outbf[(size_t)m * ldw + n] = (bf16)val;
  }
}

// ---------------- layernorm (no affine) f32 -> bf16, one wave per row ----------------
__global__ void k_ln(const float* __restrict__ x, bf16* __restrict__ out, int M) {
  int wid = (blockIdx.x * blockDim.x + threadIdx.x) >> 5;
  int lane = threadIdx.x & 31;
  if (wid >= M) return;
  const float* row = x + (size_t)wid * C_;
  float a = row[lane], b = row[lane + 32];
  float s = a + b, sq = a * a + b * b;
  for (int d = 16; d >= 1; d >>= 1) { s += __shfl_xor(s, d, 32); sq += __shfl_xor(sq, d, 32); }
  float mean = s * (1.0f / 64.0f);
  float var  = sq * (1.0f / 64.0f) - mean * mean;
  float rs = rsqrtf(var + 1e-6f);
  out[(size_t)wid * C_ + lane]      = (bf16)((a - mean) * rs);
  out[(size_t)wid * C_ + lane + 32] = (bf16)((b - mean) * rs);
}

// ---------------- flash attention, fp32 WMMA (16x16x4), dh=16, heads=4 ----------------
// mode 0: time (L=16, base=seq*16, stride=1); mode 1: space (L=512, base=seq, stride=16)
__global__ void k_attn(const float* __restrict__ qkv, bf16* __restrict__ outbf,
                       int L, int mode) {
  __shared__ float ptile[8][16][16];
  int w    = threadIdx.x >> 5;
  int lane = threadIdx.x & 31, half = lane >> 4, lm = lane & 15;
  int wid  = blockIdx.x * 8 + w;
  int tilesPer = L >> 4;
  int qt = wid % tilesPer;
  int t2 = wid / tilesPer;
  int h  = t2 & 3;
  int sq = t2 >> 2;
  int base   = mode ? sq : sq * 16;
  int stride = mode ? 16 : 1;
  int m0 = qt * 16;
  v2f qa[4];
  {
    int qrow = base + (m0 + lm) * stride;
    const float* qp = qkv + (size_t)qrow * QLD + h * 16 + half * 2;
#pragma unroll
    for (int kk = 0; kk < 4; ++kk) qa[kk] = *(const v2f*)(qp + kk * 4);  // b64 loads
  }
  float mrun[8], lrun[8];
  v8f o = {0.f, 0.f, 0.f, 0.f, 0.f, 0.f, 0.f, 0.f};
#pragma unroll
  for (int v = 0; v < 8; ++v) { mrun[v] = -3.0e38f; lrun[v] = 0.f; }

  for (int kb = 0; kb < tilesPer; ++kb) {
    v8f sc = {0.f, 0.f, 0.f, 0.f, 0.f, 0.f, 0.f, 0.f};
    int krow = base + (kb * 16 + lm) * stride;
    const float* kp = qkv + (size_t)krow * QLD + 64 + h * 16 + half * 2;
#pragma unroll
    for (int kk = 0; kk < 4; ++kk) {            // S = Q @ K^T, f32 WMMA
      v2f bv = *(const v2f*)(kp + kk * 4);      // b64 load
      sc = __builtin_amdgcn_wmma_f32_16x16x4_f32(false, qa[kk], false, bv, (short)0, sc,
                                                 false, false);
    }
#pragma unroll
    for (int v = 0; v < 8; ++v) {               // online softmax per row
      float sv = sc[v] * 0.25f;                 // 1/sqrt(dh)
      float mx = sv;
      for (int d = 1; d < 16; d <<= 1) mx = fmaxf(mx, __shfl_xor(mx, d, 32));
      float mnew  = fmaxf(mrun[v], mx);
      float alpha = __expf(mrun[v] - mnew);
      float p     = __expf(sv - mnew);
      float ps = p;
      for (int d = 1; d < 16; d <<= 1) ps += __shfl_xor(ps, d, 32);
      lrun[v] = lrun[v] * alpha + ps;
      mrun[v] = mnew;
      o[v] *= alpha;
      sc[v] = p;
    }
#pragma unroll
    for (int v = 0; v < 8; ++v) ptile[w][v + half * 8][lm] = sc[v];  // P -> LDS
    __syncthreads();
#pragma unroll
    for (int kk = 0; kk < 4; ++kk) {            // O += P @ V, f32 WMMA
      int kidx = kk * 4 + half * 2;
      v2f av = *(const v2f*)&ptile[w][lm][kidx];     // ds_load_b64
      int vr0 = base + (kb * 16 + kidx) * stride;
      int vr1 = base + (kb * 16 + kidx + 1) * stride;
      v2f bv = {qkv[(size_t)vr0 * QLD + 128 + h * 16 + lm],
                qkv[(size_t)vr1 * QLD + 128 + h * 16 + lm]};
      o = __builtin_amdgcn_wmma_f32_16x16x4_f32(false, av, false, bv, (short)0, o,
                                                false, false);
    }
    __syncthreads();
  }
#pragma unroll
  for (int v = 0; v < 8; ++v) {
    int row = base + (m0 + v + half * 8) * stride;
    outbf[(size_t)row * C_ + h * 16 + lm] = (bf16)(o[v] / lrun[v]);
  }
}

// ---------------- coords update + preds ----------------
__global__ void k_coords(float* __restrict__ coords, const float* __restrict__ delta,
                         float* __restrict__ outp, int it) {
  int i = blockIdx.x * blockDim.x + threadIdx.x;  // s*512+n
  if (i >= S_ * N_) return;
  int s = i >> 9, n = i & 511;
  int row = n * 16 + s;
  float cx = coords[i * 2 + 0] + delta[(size_t)row * FLD + 0];
  float cy = coords[i * 2 + 1] + delta[(size_t)row * FLD + 1];
  coords[i * 2 + 0] = cx;
  coords[i * 2 + 1] = cy;
  outp[(size_t)it * S_ * N_ * 2 + i * 2 + 0] = cx * 4.0f;   // * STRIDE
  outp[(size_t)it * S_ * N_ * 2 + i * 2 + 1] = cy * 4.0f;
}

// ---------------- groupnorm(df) -> bf16, one wave per row ----------------
__global__ void k_dn(const float* __restrict__ delta, const float* __restrict__ gnw,
                     const float* __restrict__ gnb, bf16* __restrict__ out, int M) {
  int wid = (blockIdx.x * blockDim.x + threadIdx.x) >> 5;
  int lane = threadIdx.x & 31;
  if (wid >= M) return;
  const float* row = delta + (size_t)wid * FLD + 2;
  float a = row[lane], b = row[lane + 32];
  float s = a + b, sq = a * a + b * b;
  for (int d = 16; d >= 1; d >>= 1) { s += __shfl_xor(s, d, 32); sq += __shfl_xor(sq, d, 32); }
  float mean = s * (1.0f / 64.0f);
  float var  = sq * (1.0f / 64.0f) - mean * mean;
  float rs = rsqrtf(var + 1e-5f);
  out[(size_t)wid * C_ + lane]      = (bf16)((a - mean) * rs * gnw[lane] + gnb[lane]);
  out[(size_t)wid * C_ + lane + 32] = (bf16)((b - mean) * rs * gnw[lane + 32] + gnb[lane + 32]);
}

extern "C" void kernel_launch(void* const* d_in, const int* in_sizes, int n_in,
                              void* d_out, int out_size, void* d_ws, size_t ws_size,
                              hipStream_t stream) {
  (void)in_sizes; (void)n_in; (void)out_size; (void)ws_size;
  const float* fmaps     = (const float*)d_in[0];
  const float* coords_in = (const float*)d_in[1];
  const float* track_in  = (const float*)d_in[2];
  auto LF = [&](int i) { return (const float*)d_in[i]; };
  const float* flowh_b = LF(3); const float* flowh_w = LF(4);
  const float* gn_b = LF(5);    const float* gn_w = LF(6);
  const float* inproj_b = LF(7); const float* inproj_w = LF(8);
  const int SPACE0 = 9, TIME0 = 41;          // per block: fc1.b/w, fc2.b/w, proj.b/w, qkv.b/w
  const float* upd_b = LF(73); const float* upd_w = LF(74);

  size_t off = 0;
  auto alloc = [&](size_t bytes) -> void* {
    void* p = (void*)((char*)d_ws + off);
    off += (bytes + 255) & ~(size_t)255;
    return p;
  };
  float* pyr1 = (float*)alloc((size_t)S_ * C_ * 32 * 32 * 4);
  float* pyr2 = (float*)alloc((size_t)S_ * C_ * 16 * 16 * 4);
  float* pyr3 = (float*)alloc((size_t)S_ * C_ * 8 * 8 * 4);
  float* coordsCur = (float*)alloc((size_t)S_ * N_ * 2 * 4);
  float* tfeatCur  = (float*)alloc((size_t)S_ * N_ * C_ * 4);
  bf16*  xbf    = (bf16*)alloc((size_t)MTOK * XK * 2);
  float* tokens = (float*)alloc((size_t)MTOK * C_ * 4);
  bf16*  lnbf   = (bf16*)alloc((size_t)MTOK * C_ * 2);
  float* qkvf   = (float*)alloc((size_t)MTOK * QLD * 4);
  bf16*  attnbf = (bf16*)alloc((size_t)MTOK * C_ * 2);
  float* hbuf   = (float*)alloc((size_t)MTOK * C_ * 4);
  bf16*  hbf    = (bf16*)alloc((size_t)MTOK * C_ * 2);
  float* deltaf = (float*)alloc((size_t)MTOK * FLD * 4);
  bf16*  tokbf  = (bf16*)alloc((size_t)MTOK * C_ * 2);
  bf16*  dnbf   = (bf16*)alloc((size_t)MTOK * C_ * 2);
  bf16*  wIn    = (bf16*)alloc((size_t)XK * 64 * 2);
  bf16 *wQKV[8], *wProj[8], *wFC1[8], *wFC2[8];
  for (int b = 0; b < 8; ++b) {
    wQKV[b] = (bf16*)alloc(64 * 192 * 2);
    wProj[b] = (bf16*)alloc(64 * 64 * 2);
    wFC1[b]  = (bf16*)alloc(64 * 64 * 2);
    wFC2[b]  = (bf16*)alloc(64 * 64 * 2);
  }
  bf16* wFlow = (bf16*)alloc(64 * FLD * 2);
  bf16* wUpd  = (bf16*)alloc(64 * 64 * 2);

  hipMemcpyAsync(coordsCur, coords_in, (size_t)S_ * N_ * 2 * 4, hipMemcpyDeviceToDevice, stream);
  hipMemcpyAsync(tfeatCur, track_in, (size_t)S_ * N_ * C_ * 4, hipMemcpyDeviceToDevice, stream);

  k_avgpool<<<(S_ * C_ * 32 * 32 + 255) / 256, 256, 0, stream>>>(fmaps, pyr1, 32, 32);
  k_avgpool<<<(S_ * C_ * 16 * 16 + 255) / 256, 256, 0, stream>>>(pyr1, pyr2, 16, 16);
  k_avgpool<<<(S_ * C_ * 8 * 8 + 255) / 256, 256, 0, stream>>>(pyr2, pyr3, 8, 8);

  auto padw = [&](const float* src, bf16* dst, int din, int dout, int dinP, int doutP) {
    int tot = dinP * doutP;
    k_padw<<<(tot + 255) / 256, 256, 0, stream>>>(src, dst, din, dout, dinP, doutP);
  };
  padw(inproj_w, wIn, 196, 64, XK, 64);
  for (int d = 0; d < 4; ++d) {
    int tb = TIME0 + 8 * d, sb = SPACE0 + 8 * d;
    padw(LF(tb + 7), wQKV[d], 64, 192, 64, 192);
    padw(LF(tb + 5), wProj[d], 64, 64, 64, 64);
    padw(LF(tb + 1), wFC1[d], 64, 64, 64, 64);
    padw(LF(tb + 3), wFC2[d], 64, 64, 64, 64);
    padw(LF(sb + 7), wQKV[4 + d], 64, 192, 64, 192);
    padw(LF(sb + 5), wProj[4 + d], 64, 64, 64, 64);
    padw(LF(sb + 1), wFC1[4 + d], 64, 64, 64, 64);
    padw(LF(sb + 3), wFC2[4 + d], 64, 64, 64, 64);
  }
  padw(flowh_w, wFlow, 64, 66, 64, FLD);
  padw(upd_w, wUpd, 64, 64, 64, 64);

  auto gemm = [&](const bf16* A, int lda, const bf16* WT, int ldw, const float* bias,
                  int dout, const float* res, float* outf, bf16* outbf, int K, int act,
                  int tfmode) {
    int waves = (MTOK / 16) * (ldw / 16);
    k_gemm<<<waves / 8, 256, 0, stream>>>(A, lda, WT, ldw, bias, dout, res, outf, outbf,
                                          MTOK, K, act, tfmode);
  };

  auto runBlock = [&](int widx, int base, int mode) {
    k_ln<<<1024, 256, 0, stream>>>(tokens, lnbf, MTOK);
    gemm(lnbf, 64, wQKV[widx], 192, LF(base + 6), 192, nullptr, qkvf, nullptr, 64, 0, 0);
    k_attn<<<256, 256, 0, stream>>>(qkvf, attnbf, mode ? 512 : 16, mode);
    gemm(attnbf, 64, wProj[widx], 64, LF(base + 4), 64, tokens, tokens, nullptr, 64, 0, 0);
    k_ln<<<1024, 256, 0, stream>>>(tokens, lnbf, MTOK);
    gemm(lnbf, 64, wFC1[widx], 64, LF(base + 0), 64, nullptr, hbuf, hbf, 64, 1, 0);
    gemm(hbf, 64, wFC2[widx], 64, LF(base + 2), 64, tokens, tokens, nullptr, 64, 0, 0);
  };

  for (int it = 0; it < 6; ++it) {
    k_corr<<<1024, 256, 0, stream>>>(coordsCur, tfeatCur, fmaps, pyr1, pyr2, pyr3, xbf);
    k_buildx<<<MTOK / 256, 256, 0, stream>>>(coordsCur, tfeatCur, xbf);
    gemm(xbf, XK, wIn, 64, inproj_b, 64, nullptr, tokens, nullptr, XK, 0, 0);
    for (int d = 0; d < 4; ++d) {
      runBlock(d, TIME0 + 8 * d, 0);       // time attention
      runBlock(4 + d, SPACE0 + 8 * d, 1);  // space attention
    }
    k_f2bf<<<(MTOK * C_) / 256, 256, 0, stream>>>(tokens, tokbf, MTOK * C_);
    gemm(tokbf, 64, wFlow, FLD, flowh_b, 66, nullptr, deltaf, nullptr, 64, 0, 0);
    k_coords<<<(S_ * N_) / 256, 256, 0, stream>>>(coordsCur, deltaf, (float*)d_out, it);
    k_dn<<<1024, 256, 0, stream>>>(deltaf, gn_w, gn_b, dnbf, MTOK);
    gemm(dnbf, 64, wUpd, 64, upd_b, 64, tfeatCur, tfeatCur, nullptr, 64, 2, 1);
  }
}
